// AttentionLayer_25933012533355
// MI455X (gfx1250) — matrix operand order repaired and checked by reading, not compiled, not run on hardware
//
#include <hip/hip_runtime.h>

#define IN_CH   10
#define OUT_CH  11
#define NROWS   8192
#define EPS     1e-5f
#define WAVES_PER_BLOCK 2

typedef __attribute__((ext_vector_type(16))) _Float16 v16h;
typedef __attribute__((ext_vector_type(8)))  _Float16 v8h_t;
typedef __attribute__((ext_vector_type(8)))  float    v8f;

union AFRAG { v16h v; v8h_t h[2]; };
union BFRAG { v16h v; unsigned u[8]; };
union QKBUF { v16h v[2]; _Float16 f[32]; };

// slot permutation for 32-blocks: memory position m holds logical index
// [0..7 -> 0..7, 8..15 -> 16..23, 16..23 -> 8..15, 24..31 -> 24..31]
// (involution: pos(d) = d ^ 24 for d in [8,24), else d)
__device__ __forceinline__ int slot32(int d) {
    return d ^ ((((d >> 3) ^ (d >> 4)) & 1) ? 24 : 0);
}

// ---------------------------------------------------------------------------
// Kernel 1: fused LayerNorm + Linear.
//  qh: [8192][32] f16, natural dim order, pre-scaled by IN_CH^-0.5 * log2(e),
//      padded dims 11..31 = 0.  (used as B-fragment of Q^T: contiguous 16 dims)
//  kh: [8192][32] f16, natural dim order, pads 0.
//      (used as A-fragment of K: one contiguous 32B load per lane)
//  vt: [16][8192] f16 transposed, keys in natural order,
//      out-ch rows 11..15 = 0. (used as A-fragment of V^T)
// ---------------------------------------------------------------------------
__global__ __launch_bounds__(256) void qkv_kernel(
    const float* __restrict__ x, const float* __restrict__ gamma,
    const float* __restrict__ beta, const float* __restrict__ W,
    _Float16* __restrict__ qh, _Float16* __restrict__ kh,
    _Float16* __restrict__ vt)
{
    int row = blockIdx.x * blockDim.x + threadIdx.x;
    if (row >= NROWS) return;

    float xv[IN_CH];
    float mu = 0.f;
#pragma unroll
    for (int i = 0; i < IN_CH; ++i) { xv[i] = x[row * IN_CH + i]; mu += xv[i]; }
    mu *= (1.0f / IN_CH);
    float var = 0.f;
#pragma unroll
    for (int i = 0; i < IN_CH; ++i) { float d = xv[i] - mu; var += d * d; }
    var *= (1.0f / IN_CH);
    float rstd = rsqrtf(var + EPS);

    float h[IN_CH];
#pragma unroll
    for (int i = 0; i < IN_CH; ++i)
        h[i] = (xv[i] - mu) * rstd * gamma[i] + beta[i];

    // IN_CH^-0.5 (attention scale) * log2(e) (exp2 domain for softmax)
    const float qscale = 0.31622776601683794f * 1.4426950408889634f;

    QKBUF qb_, kb_;
#pragma unroll
    for (int m = 0; m < 32; ++m) { qb_.f[m] = (_Float16)0.f; kb_.f[m] = (_Float16)0.f; }

    const int r5 = row & 31;
    const int rowperm = (row & ~31) | r5;

#pragma unroll
    for (int j = 0; j < 3 * OUT_CH; ++j) {
        float acc = 0.f;
#pragma unroll
        for (int i = 0; i < IN_CH; ++i) acc += h[i] * W[j * IN_CH + i];
        if (j < OUT_CH)
            qb_.f[j] = (_Float16)(acc * qscale);
        else if (j < 2 * OUT_CH)
            kb_.f[j - OUT_CH] = (_Float16)acc;
        else
            vt[(size_t)(j - 2 * OUT_CH) * NROWS + rowperm] = (_Float16)acc;
    }
    *(v16h*)(qh + (size_t)row * 32)      = qb_.v[0];
    *(v16h*)(qh + (size_t)row * 32 + 16) = qb_.v[1];
    *(v16h*)(kh + (size_t)row * 32)      = kb_.v[0];
    *(v16h*)(kh + (size_t)row * 32 + 16) = kb_.v[1];
#pragma unroll
    for (int c = OUT_CH; c < 16; ++c)
        vt[(size_t)c * NROWS + rowperm] = (_Float16)0.f;
}

// ---------------------------------------------------------------------------
// Kernel 2: fused flash attention, transposed formulation, LDS-free.
//   S^T = K x Q^T   (C-layout: VGPR g = key row, lane = query col)
//   -> per-query max/sum are per-lane scalars (+ one xor-16 combine)
//   -> P^T B-fragment built in-register: cvt_pk to f16 pairs + half-swap
//   O^T = V^T x P^T accumulated in f32, normalized and stored at the end.
//   exp2 via raw v_exp_f32 (args are <= 0; sub-denormal probs flush to 0,
//   which is exactly right for softmax accumulation).
// ---------------------------------------------------------------------------
__global__ __launch_bounds__(WAVES_PER_BLOCK * 32) void attn_kernel(
    const _Float16* __restrict__ qh, const _Float16* __restrict__ kh,
    const _Float16* __restrict__ vt, float* __restrict__ out)
{
    const int lane  = threadIdx.x & 31;
    const int wave  = threadIdx.x >> 5;
    const int tile  = blockIdx.x * WAVES_PER_BLOCK + wave;   // 0..511
    const int qbase = tile * 16;
    const int col   = lane & 15;
    const int hi    = lane >> 4;

    // Q^T B-fragment: lane -> query qbase+col, 16 contiguous dims (16*hi..)
    v16h qb = *(const v16h*)(qh + (size_t)(qbase + col) * 32 + hi * 16);

    v8f O = {};                  // O^T: VGPR g = out-ch g+8*hi, lane = query
    float m_run = -__builtin_inff();
    float l_run = 0.f;

    for (int kb = 0; kb < NROWS; kb += 32) {
        if (kb + 32 < NROWS) {
            __builtin_prefetch(kh + (size_t)(kb + 32) * 32, 0, 1);
            __builtin_prefetch(vt + (size_t)col * NROWS + kb + 32, 0, 1);
        }

        // K A-fragments (natural dim order -> single 32B load per lane)
        v16h ka0 = *(const v16h*)(kh + (size_t)(kb + col) * 32      + hi * 16);
        v16h ka1 = *(const v16h*)(kh + (size_t)(kb + 16 + col) * 32 + hi * 16);

        v8f zero = {};
        v8f st0 = __builtin_amdgcn_wmma_f32_16x16x32_f16(
            false, ka0, false, qb, (short)0, zero, false, false);
        v8f st1 = __builtin_amdgcn_wmma_f32_16x16x32_f16(
            false, ka1, false, qb, (short)0, zero, false, false);

        // per-query chunk max: in-lane tree + one half-swap
        float cm = fmaxf(st0[0], st1[0]);
#pragma unroll
        for (int g = 1; g < 8; ++g) cm = fmaxf(cm, fmaxf(st0[g], st1[g]));
        cm = fmaxf(cm, __shfl_xor(cm, 16));

        float mn = fmaxf(m_run, cm);
        float sc = __builtin_amdgcn_exp2f(m_run - mn);  // log2 domain
        m_run = mn;
        l_run *= sc;
#pragma unroll
        for (int g = 0; g < 8; ++g) O[g] *= sc;

        // exp + pack to f16 pairs (keys 2j,2j+1 of this lane's half)
        unsigned pk0[4], pk1[4];
        float lacc = 0.f;
#pragma unroll
        for (int j = 0; j < 4; ++j) {
            float p0 = __builtin_amdgcn_exp2f(st0[2 * j]     - mn);
            float p1 = __builtin_amdgcn_exp2f(st0[2 * j + 1] - mn);
            float p2 = __builtin_amdgcn_exp2f(st1[2 * j]     - mn);
            float p3 = __builtin_amdgcn_exp2f(st1[2 * j + 1] - mn);
            lacc += (p0 + p1) + (p2 + p3);
            auto a = __builtin_amdgcn_cvt_pkrtz(p0, p1);
            auto b = __builtin_amdgcn_cvt_pkrtz(p2, p3);
            __builtin_memcpy(&pk0[j], &a, 4);
            __builtin_memcpy(&pk1[j], &b, 4);
        }
        l_run += lacc;

        // half-swap packed pairs and assemble P^T B-fragment:
        // lane hi=0: slots = keys kb+0..15  = [own st0 | swapped st0]
        // lane hi=1: slots = keys kb+16..31 = [swapped st1 | own st1]
        unsigned sw0[4], sw1[4];
#pragma unroll
        for (int j = 0; j < 4; ++j) {
            sw0[j] = __shfl_xor(pk0[j], 16);
            sw1[j] = __shfl_xor(pk1[j], 16);
        }
        BFRAG pb;
#pragma unroll
        for (int j = 0; j < 4; ++j) {
            pb.u[j]     = hi ? sw1[j] : pk0[j];
            pb.u[4 + j] = hi ? pk1[j] : sw0[j];
        }

        // V^T A-fragment (natural-order keys -> single 32B load per lane)
        v16h va = *(const v16h*)(vt + (size_t)col * NROWS + kb + hi * 16);

        O = __builtin_amdgcn_wmma_f32_16x16x32_f16(
            false, va, false, pb.v, (short)0, O, false, false);
    }

    // combine lane-pair partial sums, normalize, store O^T
    float l_tot = l_run + __shfl_xor(l_run, 16);
    float inv = 1.0f / l_tot;
    const size_t orow = (size_t)(qbase + col) * OUT_CH;
#pragma unroll
    for (int g = 0; g < 8; ++g) {
        int oc = g + 8 * hi;
        if (oc < OUT_CH) out[orow + oc] = O[g] * inv;
    }
}

// ---------------------------------------------------------------------------
extern "C" void kernel_launch(void* const* d_in, const int* in_sizes, int n_in,
                              void* d_out, int out_size, void* d_ws, size_t ws_size,
                              hipStream_t stream) {
    const float* x     = (const float*)d_in[0];
    const float* gamma = (const float*)d_in[1];
    const float* beta  = (const float*)d_in[2];
    const float* W     = (const float*)d_in[3];
    float* out = (float*)d_out;

    char* ws = (char*)d_ws;
    _Float16* qh = (_Float16*)(ws);                  // 512 KB
    _Float16* kh = (_Float16*)(ws + 512 * 1024);     // 512 KB
    _Float16* vt = (_Float16*)(ws + 1024 * 1024);    // 256 KB

    qkv_kernel<<<NROWS / 256, 256, 0, stream>>>(x, gamma, beta, W, qh, kh, vt);

    int tiles  = NROWS / 16;                          // 512 query tiles
    int blocks = tiles / WAVES_PER_BLOCK;             // 256 blocks x 64 threads
    attn_kernel<<<blocks, WAVES_PER_BLOCK * 32, 0, stream>>>(qh, kh, vt, out);
}
